// MultiHeadAttention_5446018531676
// MI455X (gfx1250) — matrix operand order repaired and checked by reading, not compiled, not run on hardware
//
#include <hip/hip_runtime.h>
#include <math.h>

typedef __attribute__((ext_vector_type(16))) _Float16 v16h;
typedef __attribute__((ext_vector_type(8)))  _Float16 v8h;
typedef __attribute__((ext_vector_type(8)))  float    v8f;

#define D_MODEL 768
#define N_SEQ   577
#define N_PAD   608            // 19 chunks of 32 keys
#define BATCH   32
#define HEADS   12
#define HDIM    64
#define M_ROWS  (BATCH * N_SEQ)   // 18464 = 1154 * 16 exactly
#define M_TILES (M_ROWS / 16)     // 1154

// ---- fragment loaders using contiguous K order (k = 16*half + s) for both A and B ----
// A (16x32, MxK): lane<16 -> row=lane%16, K 0..15; lane>=16 -> K 16..31
__device__ __forceinline__ v16h load_a_frag(const _Float16* __restrict__ row, int half) {
  v8h lo = *(const v8h*)(row + half * 16);
  v8h hi = *(const v8h*)(row + half * 16 + 8);
  v16h r;
#pragma unroll
  for (int i = 0; i < 8; ++i) { r[i] = lo[i]; r[i + 8] = hi[i]; }
  return r;
}
// B (32x16, KxN): lane%16 = column, lane<16 holds K 0..15, lane>=16 holds K 16..31,
// packed in K order within the lane -> 16 contiguous halves (caller adds half*16).
__device__ __forceinline__ v16h load_b_frag(const _Float16* __restrict__ p) {
  v8h lo = *(const v8h*)p;
  v8h hi = *(const v8h*)(p + 8);
  v16h r;
#pragma unroll
  for (int i = 0; i < 8; ++i) { r[i] = lo[i]; r[i + 8] = hi[i]; }
  return r;
}

__global__ void cvt_f32_to_f16(const float* __restrict__ src,
                               _Float16* __restrict__ dst, int n) {
  int i = blockIdx.x * blockDim.x + threadIdx.x;
  int stride = gridDim.x * blockDim.x;
  for (; i < n; i += stride) dst[i] = (_Float16)src[i];
}

// Y[M,N] = X[M,K] @ W[N,K]^T + bias[N]
// MODE 0: Q -> f16 [B,H,N_PAD,HDIM]   MODE 1: K -> same
// MODE 2: V -> f16 [B,H,HDIM,N_PAD] (transposed)
// MODE 3: out-projection -> f32 [M_ROWS, D_MODEL]
template <int MODE>
__global__ __launch_bounds__(128)
void gemm_wmma_kernel(const _Float16* __restrict__ X,
                      const _Float16* __restrict__ W,
                      const float* __restrict__ bias,
                      _Float16* __restrict__ out16,
                      float* __restrict__ out32) {
  const int lane = threadIdx.x & 31;
  const int wave = threadIdx.x >> 5;
  const int half = lane >> 4;
  const int lidx = lane & 15;
  const int mtile = blockIdx.x * 4 + wave;
  if (mtile >= M_TILES) return;            // wave-uniform; EXEC stays full
  const int nbase = blockIdx.y * 64;

  v8f acc[4];
#pragma unroll
  for (int t = 0; t < 4; ++t) {
    float bv = bias[nbase + t * 16 + lidx];
#pragma unroll
    for (int i = 0; i < 8; ++i) acc[t][i] = bv;
  }

  const _Float16* xrow = X + (size_t)(mtile * 16 + lidx) * D_MODEL;
  for (int kk = 0; kk < D_MODEL; kk += 32) {
    v16h afrag = load_a_frag(xrow + kk, half);
#pragma unroll
    for (int t = 0; t < 4; ++t) {
      const _Float16* wrow =
          W + (size_t)(nbase + t * 16 + lidx) * D_MODEL + kk + half * 16;
      v16h bfrag = load_b_frag(wrow);
      acc[t] = __builtin_amdgcn_wmma_f32_16x16x32_f16(
          false, afrag, false, bfrag, (short)0, acc[t], false, false);
    }
  }

#pragma unroll
  for (int t = 0; t < 4; ++t) {
    const int nc = nbase + t * 16 + lidx;
#pragma unroll
    for (int i = 0; i < 8; ++i) {
      const int m = mtile * 16 + i + 8 * half;
      const float v = acc[t][i];
      if (MODE == 3) {
        out32[(size_t)m * D_MODEL + nc] = v;
      } else {
        const int b = m / N_SEQ;
        const int q = m - b * N_SEQ;
        const int h = nc >> 6;
        const int hd = nc & 63;
        if (MODE == 2)
          out16[((size_t)(b * HEADS + h) * HDIM + hd) * N_PAD + q] = (_Float16)v;
        else
          out16[((size_t)(b * HEADS + h) * N_PAD + q) * HDIM + hd] = (_Float16)v;
      }
    }
  }
}

// Flash attention, transposed formulation: S^T = K @ Q^T, O^T = V^T @ P^T.
// Anti-causal mask (allowed: k >= q): chunks entirely below the q-tile are
// skipped; only the diagonal and sequence-end chunks apply per-element masks.
// Running max starts at 0 so masked entries underflow exp() without guards.
// K fragments are software-pipelined one chunk ahead (loop unrolled x2 so the
// prefetch registers ping-pong instead of being copied). One wave per block.
// Q,K: [B,H,N_PAD,HDIM] f16.  Vt: [B,H,HDIM,N_PAD] f16.  Aout: [B,N_SEQ,D_MODEL] f16.
__global__ __launch_bounds__(32, 1)
void attention_kernel(const _Float16* __restrict__ Q,
                      const _Float16* __restrict__ K,
                      const _Float16* __restrict__ Vt,
                      _Float16* __restrict__ Aout) {
  const int lane = threadIdx.x & 31;
  const int half = lane >> 4;
  const int lidx = lane & 15;
  const int qb = blockIdx.x * 16;
  const int h = blockIdx.y;
  const int b = blockIdx.z;
  const float scale = 0.03608439182435161f;  // 1/sqrt(768) (faithful quirk)

  const size_t head = (size_t)(b * HEADS + h);
  const _Float16* Qh = Q + head * N_PAD * HDIM;
  const _Float16* Kh = K + head * N_PAD * HDIM;
  const _Float16* Vh = Vt + head * HDIM * N_PAD;

  // Q^T as B-operand (d x q): lane owns q column qb+lidx, d contiguous.
  const _Float16* qrow = Qh + (size_t)(qb + lidx) * HDIM;
  const v16h qfB0 = load_b_frag(qrow + half * 16);        // d 0..31
  const v16h qfB1 = load_b_frag(qrow + 32 + half * 16);   // d 32..63
  const int q = qb + lidx;
  const int lim = N_SEQ - q;       // allowed: 0 <= (k - q) < lim  (unsigned check)
  const int kstart = qb & ~31;     // first chunk containing any allowed key

  v8f o[4];  // O^T tiles: row d = t*16 + i + 8*half, col q = lidx
#pragma unroll
  for (int t = 0; t < 4; ++t)
#pragma unroll
    for (int i = 0; i < 8; ++i) o[t][i] = 0.0f;
  float mrun = 0.0f, lrun = 0.0f;  // m >= 0 keeps every exp() arg <= 0

  // K fragments for the first (diagonal) chunk
  const _Float16* kb0 = Kh + (size_t)(kstart + lidx) * HDIM;
  v16h kfa0 = load_a_frag(kb0, half);
  v16h kfa1 = load_a_frag(kb0 + 32, half);
  v16h kfb0 = load_a_frag(kb0 + 16 * HDIM, half);
  v16h kfb1 = load_a_frag(kb0 + 16 * HDIM + 32, half);

#pragma unroll 2
  for (int k0 = kstart; k0 < N_PAD; k0 += 32) {
    // V fragments for this chunk (only needed by the tail PV WMMAs)
    v16h vf0 = load_a_frag(Vh + (size_t)(0 * 16 + lidx) * N_PAD + k0, half);
    v16h vf1 = load_a_frag(Vh + (size_t)(1 * 16 + lidx) * N_PAD + k0, half);
    v16h vf2 = load_a_frag(Vh + (size_t)(2 * 16 + lidx) * N_PAD + k0, half);
    v16h vf3 = load_a_frag(Vh + (size_t)(3 * 16 + lidx) * N_PAD + k0, half);

    // ---- S^T(32k x 16q): A = K rows (keys x d), B = Q^T ----
    v8f sa; v8f sb;
#pragma unroll
    for (int i = 0; i < 8; ++i) { sa[i] = 0.0f; sb[i] = 0.0f; }
    sa = __builtin_amdgcn_wmma_f32_16x16x32_f16(false, kfa0, false, qfB0,
                                                (short)0, sa, false, false);
    sa = __builtin_amdgcn_wmma_f32_16x16x32_f16(false, kfa1, false, qfB1,
                                                (short)0, sa, false, false);
    sb = __builtin_amdgcn_wmma_f32_16x16x32_f16(false, kfb0, false, qfB0,
                                                (short)0, sb, false, false);
    sb = __builtin_amdgcn_wmma_f32_16x16x32_f16(false, kfb1, false, qfB1,
                                                (short)0, sb, false, false);

    // ---- prefetch next chunk's K into fresh registers (clamped, no branch) ----
    const int kn = (k0 + 32 < N_PAD) ? (k0 + 32) : k0;
    const _Float16* knb = Kh + (size_t)(kn + lidx) * HDIM;
    kfa0 = load_a_frag(knb, half);
    kfa1 = load_a_frag(knb + 32, half);
    kfb0 = load_a_frag(knb + 16 * HDIM, half);
    kfb1 = load_a_frag(knb + 16 * HDIM + 32, half);

    // ---- scale (+ mask only on diagonal / tail chunks; uniform branch) ----
    float pA[8], pB[8];
    float t = -3.0e38f;
    if (k0 == kstart || k0 + 32 > N_SEQ) {
      const int rel = k0 - q + 8 * half;
#pragma unroll
      for (int i = 0; i < 8; ++i) {
        const int ra = rel + i;                  // group-a: k - q
        const int rb = ra + 16;                  // group-b: k - q
        pA[i] = ((unsigned)ra < (unsigned)lim) ? sa[i] * scale : -3.0e38f;
        pB[i] = ((unsigned)rb < (unsigned)lim) ? sb[i] * scale : -3.0e38f;
        t = fmaxf(t, fmaxf(pA[i], pB[i]));
      }
    } else {
#pragma unroll
      for (int i = 0; i < 8; ++i) {
        pA[i] = sa[i] * scale;
        pB[i] = sb[i] * scale;
        t = fmaxf(t, fmaxf(pA[i], pB[i]));
      }
    }

    // ---- online softmax: keys in-lane (i, half), q is the lane ----
    t = fmaxf(t, __shfl_xor(t, 16, 32));         // combine the two halves
    const float mnew = fmaxf(mrun, t);
    const float a = __expf(mrun - mnew);
    mrun = mnew;
    float s = 0.0f;
#pragma unroll
    for (int i = 0; i < 8; ++i) {
      pA[i] = __expf(pA[i] - mnew);              // masked -> exp(-huge) = 0
      pB[i] = __expf(pB[i] - mnew);
      s += pA[i] + pB[i];
    }
    s += __shfl_xor(s, 16, 32);
    lrun = lrun * a + s;
#pragma unroll
    for (int tt = 0; tt < 4; ++tt)
#pragma unroll
      for (int i = 0; i < 8; ++i) o[tt][i] *= a;

    // ---- P^T -> B-operand via one half-swap: each lane sends what its
    // partner needs (half==0 sends groupB, half==1 sends groupA).
    // lane<16 frag = [own groupA, partner groupA]
    // lane>=16 frag = [partner groupB, own groupB]
    union Pk { v8h h; unsigned u[4]; };
    Pk au, bu, snd, swp;
#pragma unroll
    for (int i = 0; i < 8; ++i) { au.h[i] = (_Float16)pA[i]; bu.h[i] = (_Float16)pB[i]; }
#pragma unroll
    for (int j = 0; j < 4; ++j) snd.u[j] = half ? au.u[j] : bu.u[j];
#pragma unroll
    for (int j = 0; j < 4; ++j) swp.u[j] = __shfl_xor(snd.u[j], 16, 32);
    Pk plo, phi;
#pragma unroll
    for (int j = 0; j < 4; ++j) {
      plo.u[j] = half ? swp.u[j] : au.u[j];      // k 0..7  | k 16..23
      phi.u[j] = half ? bu.u[j]  : swp.u[j];     // k 8..15 | k 24..31
    }
    v16h pf;
#pragma unroll
    for (int i = 0; i < 8; ++i) { pf[i] = plo.h[i]; pf[i + 8] = phi.h[i]; }

    // ---- O^T(64d x 16q) += V^T(d x 32k) @ P^T(32k x 16q) ----
    o[0] = __builtin_amdgcn_wmma_f32_16x16x32_f16(false, vf0, false, pf,
                                                  (short)0, o[0], false, false);
    o[1] = __builtin_amdgcn_wmma_f32_16x16x32_f16(false, vf1, false, pf,
                                                  (short)0, o[1], false, false);
    o[2] = __builtin_amdgcn_wmma_f32_16x16x32_f16(false, vf2, false, pf,
                                                  (short)0, o[2], false, false);
    o[3] = __builtin_amdgcn_wmma_f32_16x16x32_f16(false, vf3, false, pf,
                                                  (short)0, o[3], false, false);
  }

  // ---- normalize + store: lane owns one q row, d contiguous -> vector stores ----
  const float linv = 1.0f / lrun;
  if (q < N_SEQ) {
    _Float16* orow = Aout + (size_t)(b * N_SEQ + q) * D_MODEL + h * HDIM + 8 * half;
#pragma unroll
    for (int tt = 0; tt < 4; ++tt) {
      v8h ov;
#pragma unroll
      for (int i = 0; i < 8; ++i) ov[i] = (_Float16)(o[tt][i] * linv);
      *(v8h*)(orow + tt * 16) = ov;
    }
  }
}

extern "C" void kernel_launch(void* const* d_in, const int* in_sizes, int n_in,
                              void* d_out, int out_size, void* d_ws, size_t ws_size,
                              hipStream_t stream) {
  (void)in_sizes; (void)n_in; (void)out_size; (void)ws_size;
  const float* x  = (const float*)d_in[0];
  const float* Wq = (const float*)d_in[1];
  const float* bq = (const float*)d_in[2];
  const float* Wk = (const float*)d_in[3];
  const float* bk = (const float*)d_in[4];
  const float* Wv = (const float*)d_in[5];
  const float* bv = (const float*)d_in[6];
  const float* Wo = (const float*)d_in[7];
  const float* bo = (const float*)d_in[8];
  float* out = (float*)d_out;

  const size_t szX = (size_t)M_ROWS * D_MODEL;             // 14,180,352 elems
  const size_t szW = (size_t)D_MODEL * D_MODEL;            //    589,824 elems
  const size_t szH = (size_t)BATCH * HEADS * N_PAD * HDIM; // 14,942,208 elems

  _Float16* Xh  = (_Float16*)d_ws;       // reused as attention output later
  _Float16* Wqh = Xh + szX;
  _Float16* Wkh = Wqh + szW;
  _Float16* Wvh = Wkh + szW;
  _Float16* Woh = Wvh + szW;
  _Float16* Qh  = Woh + szW;
  _Float16* Kh  = Qh + szH;
  _Float16* Vth = Kh + szH;
  _Float16* Ah  = Xh;   // alias: x is dead after QKV GEMMs

  // 1) f32 -> f16 conversions
  cvt_f32_to_f16<<<2048, 256, 0, stream>>>(x, Xh, (int)szX);
  cvt_f32_to_f16<<<1024, 256, 0, stream>>>(Wq, Wqh, (int)szW);
  cvt_f32_to_f16<<<1024, 256, 0, stream>>>(Wk, Wkh, (int)szW);
  cvt_f32_to_f16<<<1024, 256, 0, stream>>>(Wv, Wvh, (int)szW);
  cvt_f32_to_f16<<<1024, 256, 0, stream>>>(Wo, Woh, (int)szW);

  // 2) Q/K/V projections (WMMA)
  dim3 ggrid((M_TILES + 3) / 4, D_MODEL / 64);
  gemm_wmma_kernel<0><<<ggrid, 128, 0, stream>>>(Xh, Wqh, bq, Qh, nullptr);
  gemm_wmma_kernel<1><<<ggrid, 128, 0, stream>>>(Xh, Wkh, bk, Kh, nullptr);
  gemm_wmma_kernel<2><<<ggrid, 128, 0, stream>>>(Xh, Wvh, bv, Vth, nullptr);

  // 3) masked flash attention (WMMA, triangular chunk skipping)
  attention_kernel<<<dim3(N_PAD / 16 - 1, HEADS, BATCH), 32, 0, stream>>>(Qh, Kh, Vth, Ah);

  // 4) output projection -> f32 (WMMA)
  gemm_wmma_kernel<3><<<ggrid, 128, 0, stream>>>(Ah, Woh, bo, nullptr, out);
}
